// SparseMoE_20426864459936
// MI455X (gfx1250) — matrix operand order repaired and checked by reading, not compiled, not run on hardware
//
#include <hip/hip_runtime.h>

// ---- problem constants (from reference) ----
#define NB 4
#define NT 4096
#define ND 512
#define NE 16
#define NH 2048           // 4*D
#define NN (NB*NT)        // 16384 tokens
#define CAP 2048          // N*TOPK/E*CAP_FACTOR
#define TM 64             // rows per FFN workgroup
#define KC 64             // hidden chunk
#define NCHUNK (NH/KC)    // 32

typedef __attribute__((ext_vector_type(16))) __bf16    v16bf;
typedef __attribute__((ext_vector_type(8)))  float     v8f;
typedef __attribute__((ext_vector_type(4)))  unsigned  v4u;
typedef __attribute__((ext_vector_type(8)))  int       v8i;
typedef __attribute__((ext_vector_type(4)))  int       v4i;

union FragAB { v16bf v; uint4 q[2]; };
union FragC  { v8f v; float f[8]; };

__device__ __forceinline__ unsigned short f2bf(float f) {
    union { float f; unsigned u; } c; c.f = f;
    unsigned u = c.u;
    u += 0x7FFFu + ((u >> 16) & 1u);      // round-to-nearest-even
    return (unsigned short)(u >> 16);
}

// ------------------------------------------------------------------
// Kernel 0: zero the output (it is accumulated into)
// ------------------------------------------------------------------
__global__ void zero_out_kernel(float4* __restrict__ p, int n4) {
    int i = blockIdx.x * 256 + threadIdx.x;
    if (i < n4) p[i] = make_float4(0.f, 0.f, 0.f, 0.f);
}

// ------------------------------------------------------------------
// Prep: x -> bf16 (one-time, elementwise)
// ------------------------------------------------------------------
__global__ void prep_x_kernel(const float4* __restrict__ x4, ushort4* __restrict__ o4, int n4) {
    int i = blockIdx.x * 256 + threadIdx.x;
    if (i < n4) {
        float4 v = x4[i];
        ushort4 o;
        o.x = f2bf(v.x); o.y = f2bf(v.y); o.z = f2bf(v.z); o.w = f2bf(v.w);
        o4[i] = o;
    }
}

// ------------------------------------------------------------------
// Prep: W1[E][D][H] -> W1T bf16 [E][H][D]   (chunk ci = rows h c0..c0+63,
// fully contiguous 64KB per chunk). Tiled 64x64 transpose via LDS.
// grid = (NH/64, ND/64, NE), block = 256
// ------------------------------------------------------------------
__global__ __launch_bounds__(256) void prep_w1t_kernel(
    const float* __restrict__ W1, unsigned short* __restrict__ w1t) {
    __shared__ unsigned short tile[64][68];
    const int h0 = blockIdx.x * 64, d0 = blockIdx.y * 64, e = blockIdx.z;
    for (int i = threadIdx.x; i < 4096; i += 256) {
        int dr = i >> 6, hc = i & 63;
        tile[dr][hc] = f2bf(W1[((size_t)e * ND + d0 + dr) * NH + h0 + hc]);
    }
    __syncthreads();
    for (int i = threadIdx.x; i < 4096; i += 256) {
        int hr = i >> 6, dc = i & 63;
        w1t[((size_t)e * NH + h0 + hr) * ND + d0 + dc] = tile[dc][hr];
    }
}

// ------------------------------------------------------------------
// Prep: W2[E][H][D] -> W2T bf16 blocked [E][NCHUNK][ND][KC]
// (chunk ci contiguous 64KB: [n][k_local]).  grid = (ND/64, NCHUNK, NE)
// ------------------------------------------------------------------
__global__ __launch_bounds__(256) void prep_w2t_kernel(
    const float* __restrict__ W2, unsigned short* __restrict__ w2t) {
    __shared__ unsigned short tile[64][68];
    const int n0 = blockIdx.x * 64, ci = blockIdx.y, e = blockIdx.z;
    for (int i = threadIdx.x; i < 4096; i += 256) {
        int kr = i >> 6, nc = i & 63;
        tile[kr][nc] = f2bf(W2[((size_t)e * NH + ci * KC + kr) * ND + n0 + nc]);
    }
    __syncthreads();
    for (int i = threadIdx.x; i < 4096; i += 256) {
        int nr = i >> 6, kc = i & 63;
        w2t[(((size_t)e * NCHUNK + ci) * ND + n0 + nr) * KC + kc] = tile[kc][nr];
    }
}

// ------------------------------------------------------------------
// Kernel 1: router — noisy logits, top-2, softmax gates
// ------------------------------------------------------------------
__global__ __launch_bounds__(256) void router_kernel(
    const float* __restrict__ x, const float* __restrict__ noise,
    const float* __restrict__ Wl, const float* __restrict__ bl,
    const float* __restrict__ Wn, const float* __restrict__ bn,
    int* __restrict__ ids, float* __restrict__ gts)
{
    __shared__ float xs[16][ND];
    __shared__ float nz[16][NE];
    const int t0 = blockIdx.x * 16;

    for (int i = threadIdx.x; i < 16 * ND; i += 256) {
        int r = i >> 9, c = i & (ND - 1);
        xs[r][c] = x[(size_t)(t0 + r) * ND + c];
    }
    __syncthreads();

    const int tt = threadIdx.x >> 4;
    const int e  = threadIdx.x & 15;
    float accl = 0.f, accn = 0.f;
    for (int d = 0; d < ND; ++d) {
        float xv = xs[tt][d];
        accl = fmaf(xv, Wl[d * NE + e], accl);
        accn = fmaf(xv, Wn[d * NE + e], accn);
    }
    float logit = accl + bl[e];
    float z = accn + bn[e];
    float sp = (z > 20.f) ? z : log1pf(expf(z));         // softplus
    nz[tt][e] = logit + noise[(size_t)(t0 + tt) * NE + e] * sp;
    __syncthreads();

    if (threadIdx.x < 16) {
        const int tk = threadIdx.x;
        float b0 = -3.4e38f; int i0 = 0;
        for (int j = 0; j < NE; ++j) { float v = nz[tk][j]; if (v > b0) { b0 = v; i0 = j; } }
        float s1 = -3.4e38f; int i1 = 0;
        for (int j = 0; j < NE; ++j) {
            if (j == i0) continue;
            float v = nz[tk][j]; if (v > s1) { s1 = v; i1 = j; }
        }
        float e1 = expf(s1 - b0);
        float inv = 1.f / (1.f + e1);
        size_t n = (size_t)(t0 + tk);
        ids[n * 2 + 0] = i0; ids[n * 2 + 1] = i1;
        gts[n * 2 + 0] = inv; gts[n * 2 + 1] = e1 * inv;
    }
}

// ------------------------------------------------------------------
// Kernel 2: deterministic capacity routing (ballot prefix sums)
// ------------------------------------------------------------------
__global__ __launch_bounds__(1024) void assign_kernel(
    const int* __restrict__ ids, const float* __restrict__ gts,
    int* __restrict__ sel, float* __restrict__ gsel)
{
    const int e = blockIdx.x;
    __shared__ unsigned wtot[32];
    __shared__ unsigned basesh;
    if (threadIdx.x == 0) basesh = 0;
    __syncthreads();

    const int lane = threadIdx.x & 31;
    const int wid  = threadIdx.x >> 5;

    for (int it = 0; it < NN / 1024; ++it) {
        int n = it * 1024 + threadIdx.x;
        int a0 = ids[n * 2], a1 = ids[n * 2 + 1];
        bool memb = (a0 == e) || (a1 == e);
        unsigned mask = __builtin_amdgcn_ballot_w32(memb);
        unsigned pre  = __builtin_popcount(mask & ((1u << lane) - 1u));
        if (lane == 31) wtot[wid] = __builtin_popcount(mask);
        __syncthreads();
        if (threadIdx.x == 0) {
            unsigned run = basesh;
            for (int w = 0; w < 32; ++w) { unsigned t = wtot[w]; wtot[w] = run; run += t; }
            basesh = run;
        }
        __syncthreads();
        unsigned slot = wtot[wid] + pre;
        if (memb && slot < CAP) {
            sel[e * CAP + slot]  = n;
            gsel[e * CAP + slot] = (a0 == e) ? gts[n * 2] : gts[n * 2 + 1];
        }
        __syncthreads();
    }
    unsigned cnt = basesh > CAP ? CAP : basesh;
    for (unsigned i = cnt + threadIdx.x; i < CAP; i += 1024) {
        sel[e * CAP + i] = 0; gsel[e * CAP + i] = 0.f;   // padded slots: gate 0
    }
}

// ------------------------------------------------------------------
// Kernel 3: fused expert FFN, bf16 WMMA + TDM-staged weights
// ------------------------------------------------------------------
#define XS_STRIDE 520   // 512 + 8 bf16 pad (row = 65 x 16B)
#define W_STRIDE  72    // 64 + 8 bf16 pad  (row = 9 x 16B)
#define OFF_TOK   0
#define OFF_GATE  256
#define OFF_XS    512
#define OFF_W1S   (OFF_XS  + TM * XS_STRIDE * 2)   // 67072
#define OFF_W2S0  (OFF_W1S + TM * XS_STRIDE * 2)   // 133632
#define W2S_BYTES (ND * W_STRIDE * 2)              // 73728
#define OFF_W2S1  (OFF_W2S0 + W2S_BYTES)           // 207360
#define OFF_HS    (OFF_W2S1 + W2S_BYTES)           // 281088
#define SMEM_BYTES (OFF_HS + TM * W_STRIDE * 2)    // 290304  (<= 320KB WGP LDS)

// TDM group1 word0: data_size=2B, pad_enable, pad_interval/amount per matrix
// W1 chunk rows = 512 elems = 256 dwords -> interval code 7; pad 4 dwords -> code 3
#define W1CTL 0x07D10000u
// W2 chunk rows = 64 elems = 32 dwords  -> interval code 4; pad 4 dwords -> code 3
#define W2CTL 0x07110000u

#if __has_builtin(__builtin_amdgcn_tensor_load_to_lds)
#define USE_TDM 1
#else
#define USE_TDM 0
#endif

#if USE_TDM
// Load one contiguous 64KB (32768 bf16) chunk into LDS via the Tensor Data
// Mover, with row padding applied on the LDS side. Issued once per wave.
// This toolchain exposes the 6-arg builtin:
//   (v4u g0, v8i g1, v4i g2, v4i g3, v8i g4, i32 cpol)
__device__ __forceinline__ void tdm_load_chunk(unsigned lds_off, const void* gptr,
                                               unsigned ctl) {
    unsigned long long ga = (unsigned long long)(size_t)gptr;
    v4u g0;
    g0[0] = 1u;                                            // count=1, user D#
    g0[1] = lds_off;                                       // LDS byte address
    g0[2] = (unsigned)ga;                                  // global_addr lo
    g0[3] = (unsigned)((ga >> 32) & 0x01FFFFFFu) | (2u << 30); // ga hi | type=2
    v8i g1;
    g1[0] = (int)ctl;                // wg_mask=0 | data_size=1(2B) | padding
    g1[1] = (int)0x80000000u;        // tensor_dim0 = 32768 (low16 @ [31:16])
    g1[2] = 0;                       // tensor_dim0 hi16 / tensor_dim1 lo16
    g1[3] = (int)0x80000000u;        // tensor_dim1 hi16 / tile_dim0 = 32768
    g1[4] = 0;                       // tile_dim1 = 0 (1D), tile_dim2 = 0
    g1[5] = 32768;                   // tensor_dim0_stride lo32
    g1[6] = 0;
    g1[7] = 0;
    v4i z4 = {0, 0, 0, 0};
    v8i z8 = {0, 0, 0, 0, 0, 0, 0, 0};
    __builtin_amdgcn_tensor_load_to_lds(g0, g1, z4, z4, z8, 0);
}
#else
// Fallback: cooperative vector copy (all 512 threads), applying the same
// row padding in software.
__device__ __forceinline__ void copy_chunk_w1(unsigned short* dst,
                                              const unsigned short* src, int tid) {
#pragma unroll
    for (int p = 0; p < 8; ++p) {
        int li = p * 512 + tid;            // 4096 x 16B units
        int row = li >> 6, unit = li & 63; // 64 rows x 64 units
        *(uint4*)&dst[row * XS_STRIDE + unit * 8] =
            *(const uint4*)&src[row * 512 + unit * 8];
    }
}
__device__ __forceinline__ void copy_chunk_w2(unsigned short* dst,
                                              const unsigned short* src, int tid) {
#pragma unroll
    for (int p = 0; p < 8; ++p) {
        int li = p * 512 + tid;
        int row = li >> 3, unit = li & 7;  // 512 rows x 8 units
        *(uint4*)&dst[row * W_STRIDE + unit * 8] =
            *(const uint4*)&src[row * 64 + unit * 8];
    }
}
#endif

__global__ __launch_bounds__(512) void ffn_kernel(
    const unsigned short* __restrict__ xbf,
    const unsigned short* __restrict__ w1t,
    const unsigned short* __restrict__ w2t,
    const float* __restrict__ b1, const float* __restrict__ b2,
    const int* __restrict__ sel, const float* __restrict__ gsel,
    float* __restrict__ out)
{
    extern __shared__ char smem[];
    int*            tokr  = (int*)(smem + OFF_TOK);
    float*          gater = (float*)(smem + OFF_GATE);
    unsigned short* xs    = (unsigned short*)(smem + OFF_XS);
    unsigned short* w1s   = (unsigned short*)(smem + OFF_W1S);
    unsigned short* hs    = (unsigned short*)(smem + OFF_HS);

    const int e    = blockIdx.y;
    const int row0 = blockIdx.x * TM;
    const int tid  = threadIdx.x;
    const int wid  = tid >> 5;
    const int lane = tid & 31;
    const int lm   = lane & 15;
    const int hi   = lane >> 4;
    const int r    = wid >> 2;
    const int c    = wid & 3;

    const float* b1e = b1 + (size_t)e * NH;
    const float* b2e = b2 + (size_t)e * ND;
    const unsigned short* w1te = w1t + (size_t)e * NH * ND;           // [NH][ND]
    const unsigned short* w2te = w2t + (size_t)e * NCHUNK * ND * KC;  // [NCHUNK][ND][KC]

    if (tid < TM) {
        tokr[tid]  = sel[e * CAP + row0 + tid];
        gater[tid] = gsel[e * CAP + row0 + tid];
    }
#if USE_TDM
    if (wid == 0) {   // one wave issues the DMA (TDM ignores EXEC)
        tdm_load_chunk(OFF_W1S,  w1te, W1CTL);
        tdm_load_chunk(OFF_W2S0, w2te, W2CTL);
    }
#endif
    __syncthreads();   // tokr visible

    // gather x rows (bf16, b128 copies) — overlaps the TDM fill
#pragma unroll
    for (int p = 0; p < 8; ++p) {
        int li = p * 512 + tid;
        int row = li >> 6, unit = li & 63;
        *(uint4*)&xs[row * XS_STRIDE + unit * 8] =
            *(const uint4*)&xbf[(size_t)tokr[row] * ND + unit * 8];
    }
#if USE_TDM
    if (wid == 0) __builtin_amdgcn_s_wait_tensorcnt((short)0);
#else
    copy_chunk_w1(w1s, w1te, tid);
    copy_chunk_w2((unsigned short*)(smem + OFF_W2S0), w2te, tid);
#endif
    __syncthreads();

    FragC acc[8];
#pragma unroll
    for (int j = 0; j < 8; ++j)
#pragma unroll
        for (int v = 0; v < 8; ++v) acc[j].f[v] = 0.f;

    for (int ci = 0; ci < NCHUNK; ++ci) {
        const int c0 = ci * KC;
        const int buf = ci & 1;
        unsigned short* w2s = (unsigned short*)(smem + (buf ? OFF_W2S1 : OFF_W2S0));

        // ---- GEMM1: h(64x64) = xs(64x512) @ W1chunk(512x64), 1 tile/wave ----
        {
            v8f hacc = {0.f,0.f,0.f,0.f,0.f,0.f,0.f,0.f};
            const unsigned short* xrow = xs  + (16 * r + lm) * XS_STRIDE;
            const unsigned short* wrow = w1s + (16 * c + lm) * XS_STRIDE;
            FragAB a[2], b[2];
            a[0].q[0] = *(const uint4*)&xrow[hi * 16];
            a[0].q[1] = *(const uint4*)&xrow[hi * 16 + 8];
            b[0].q[0] = *(const uint4*)&wrow[hi * 16];
            b[0].q[1] = *(const uint4*)&wrow[hi * 16 + 8];
#pragma unroll
            for (int kk = 0; kk < ND / 32; ++kk) {
                const int cur = kk & 1, nxt = cur ^ 1;
                if (kk < ND / 32 - 1) {      // prefetch next fragments
                    const int ka = (kk + 1) * 32 + hi * 16;
                    const int kb = (kk + 1) * 32 + hi * 16;
                    a[nxt].q[0] = *(const uint4*)&xrow[ka];
                    a[nxt].q[1] = *(const uint4*)&xrow[ka + 8];
                    b[nxt].q[0] = *(const uint4*)&wrow[kb];
                    b[nxt].q[1] = *(const uint4*)&wrow[kb + 8];
                }
                hacc = __builtin_amdgcn_wmma_f32_16x16x32_bf16(
                    false, a[cur].v, false, b[cur].v, (short)0, hacc, false, false);
            }
            const float bias = b1e[c0 + 16 * c + lm];
#pragma unroll
            for (int v = 0; v < 8; ++v) {
                int mr = 16 * r + v + hi * 8;
                float hv = hacc[v] + bias;
                hv = hv > 0.f ? hv : 0.f;
                hs[mr * W_STRIDE + 16 * c + lm] = f2bf(hv);
            }
        }
        __syncthreads();          // hs ready; w1s dead

#if USE_TDM
        if (wid == 0 && ci + 1 < NCHUNK) {   // prefetch next chunk during GEMM2
            tdm_load_chunk(OFF_W1S, w1te + (size_t)(ci + 1) * TM * ND, W1CTL);
            tdm_load_chunk(buf ? OFF_W2S0 : OFF_W2S1,
                           w2te + (size_t)(ci + 1) * ND * KC, W2CTL);
        }
#else
        if (ci + 1 < NCHUNK) {
            copy_chunk_w1(w1s, w1te + (size_t)(ci + 1) * TM * ND, tid);
            copy_chunk_w2((unsigned short*)(smem + (buf ? OFF_W2S0 : OFF_W2S1)),
                          w2te + (size_t)(ci + 1) * ND * KC, tid);
        }
#endif

        // ---- GEMM2: out(64x512) += hs(64x64) @ W2chunk(64x512), 8 tiles/wave ----
#pragma unroll
        for (int kk = 0; kk < KC / 32; ++kk) {
            FragAB a;
            const unsigned short* hrow = hs + (16 * r + lm) * W_STRIDE;
            const int ka = kk * 32 + hi * 16;
            a.q[0] = *(const uint4*)&hrow[ka];
            a.q[1] = *(const uint4*)&hrow[ka + 8];
            const int kb = kk * 32 + hi * 16;
            FragAB bb[2];
            bb[0].q[0] = *(const uint4*)&w2s[(16 * c + lm) * W_STRIDE + kb];
            bb[0].q[1] = *(const uint4*)&w2s[(16 * c + lm) * W_STRIDE + kb + 8];
#pragma unroll
            for (int j = 0; j < 8; ++j) {
                const int cur = j & 1, nxt = cur ^ 1;
                if (j < 7) {                  // prefetch next B fragment
                    const int nn = 16 * (c + 4 * (j + 1)) + lm;
                    bb[nxt].q[0] = *(const uint4*)&w2s[nn * W_STRIDE + kb];
                    bb[nxt].q[1] = *(const uint4*)&w2s[nn * W_STRIDE + kb + 8];
                }
                acc[j].v = __builtin_amdgcn_wmma_f32_16x16x32_bf16(
                    false, a.v, false, bb[cur].v, (short)0, acc[j].v, false, false);
            }
        }

#if USE_TDM
        if (wid == 0) __builtin_amdgcn_s_wait_tensorcnt((short)0);
#endif
        __syncthreads();          // next chunk's w1s/w2s ready; hs reusable
    }

    // epilogue: + b2, * gate, scatter-add
#pragma unroll
    for (int j = 0; j < 8; ++j) {
        const int nc = 16 * (c + 4 * j) + lm;
        const float bb = b2e[nc];
#pragma unroll
        for (int v = 0; v < 8; ++v) {
            int rr = 16 * r + v + hi * 8;
            float g = gater[rr];
            if (g != 0.f) {
                float val = (acc[j].f[v] + bb) * g;
                unsafeAtomicAdd(&out[(size_t)tokr[rr] * ND + nc], val);
            }
        }
    }
}

// ------------------------------------------------------------------
extern "C" void kernel_launch(void* const* d_in, const int* in_sizes, int n_in,
                              void* d_out, int out_size, void* d_ws, size_t ws_size,
                              hipStream_t stream) {
    const float* x     = (const float*)d_in[0];
    const float* noise = (const float*)d_in[1];
    const float* Wl    = (const float*)d_in[2];
    const float* bl    = (const float*)d_in[3];
    const float* Wn    = (const float*)d_in[4];
    const float* bn    = (const float*)d_in[5];
    const float* W1    = (const float*)d_in[6];
    const float* b1    = (const float*)d_in[7];
    const float* W2    = (const float*)d_in[8];
    const float* b2    = (const float*)d_in[9];
    float* out = (float*)d_out;

    char* ws = (char*)d_ws;
    int*            sel  = (int*)           (ws + 0);         // 128 KB
    float*          gsel = (float*)         (ws + 131072);    // 128 KB
    int*            ids  = (int*)           (ws + 262144);    // 128 KB
    float*          gts  = (float*)         (ws + 393216);    // 128 KB
    unsigned short* xbf  = (unsigned short*)(ws + 524288);    // 16 MB
    unsigned short* w1t  = (unsigned short*)(ws + 524288 + (size_t)16*1024*1024);  // 32 MB
    unsigned short* w2t  = (unsigned short*)(ws + 524288 + (size_t)48*1024*1024);  // 32 MB

    const int n4 = NN * ND / 4;
    zero_out_kernel<<<(n4 + 255) / 256, 256, 0, stream>>>((float4*)out, n4);
    prep_x_kernel<<<(n4 + 255) / 256, 256, 0, stream>>>(
        (const float4*)x, (ushort4*)xbf, n4);
    prep_w1t_kernel<<<dim3(NH / 64, ND / 64, NE), 256, 0, stream>>>(W1, w1t);
    prep_w2t_kernel<<<dim3(ND / 64, NCHUNK, NE), 256, 0, stream>>>(W2, w2t);

    router_kernel<<<NN / 16, 256, 0, stream>>>(x, noise, Wl, bl, Wn, bn, ids, gts);
    assign_kernel<<<NE, 1024, 0, stream>>>(ids, gts, sel, gsel);

    dim3 grid(CAP / TM, NE);
    ffn_kernel<<<grid, 512, SMEM_BYTES, stream>>>(
        xbf, w1t, w2t, b1, b2, sel, gsel, out);
}
